// GenuineAttentionHead_86148454023503
// MI455X (gfx1250) — matrix operand order repaired and checked by reading, not compiled, not run on hardware
//
#include <hip/hip_runtime.h>
#include <hip/hip_bf16.h>

#define B_  4
#define S_  2048
#define DM  1024
#define DH  64

typedef __attribute__((ext_vector_type(16))) __bf16 v16bf;
typedef __attribute__((ext_vector_type(8)))  __bf16 v8bf;
typedef __attribute__((ext_vector_type(8)))  float  v8f;
typedef __attribute__((ext_vector_type(4)))  int    i32x4;

// gfx1250 async global->LDS copy path (ASYNCcnt-tracked), guarded so the file
// still compiles on toolchains that don't expose the builtins.
#if defined(__has_builtin)
#if __has_builtin(__builtin_amdgcn_global_load_async_to_lds_b128) && \
    __has_builtin(__builtin_amdgcn_s_wait_asynccnt)
#define USE_ASYNC_LDS 1
#endif
#endif

__device__ __forceinline__ void copy16_g2lds(const __bf16* g, __bf16* l) {
#ifdef USE_ASYNC_LDS
  // Builtin signature (from hipcc diagnostic): (i32x4 AS1*, i32x4 AS3*, imm, imm)
  __builtin_amdgcn_global_load_async_to_lds_b128(
      (__attribute__((address_space(1))) i32x4*)g,
      (__attribute__((address_space(3))) i32x4*)l,
      /*offset=*/0, /*cpol=*/0);
#else
  *(v8bf*)l = *(const v8bf*)g;
#endif
}

__device__ __forceinline__ void async_copy_fence() {
#ifdef USE_ASYNC_LDS
  __builtin_amdgcn_s_wait_asynccnt(0);
#endif
}

__device__ __forceinline__ v8f wmma_bf16_f32(v16bf a, v16bf b, v8f c) {
  // D = A(16x32 bf16) * B(32x16 bf16) + C(16x16 f32)
  return __builtin_amdgcn_wmma_f32_16x16x32_bf16(
      /*neg_a=*/false, a, /*neg_b=*/false, b,
      /*c_mod=*/(short)0, c, /*reuse_a=*/false, /*reuse_b=*/false);
}

__device__ __forceinline__ float red_max16(float v) {
  v = fmaxf(v, __shfl_xor(v, 1, 32));
  v = fmaxf(v, __shfl_xor(v, 2, 32));
  v = fmaxf(v, __shfl_xor(v, 4, 32));
  v = fmaxf(v, __shfl_xor(v, 8, 32));
  return v;
}
__device__ __forceinline__ float red_sum16(float v) {
  v += __shfl_xor(v, 1, 32);
  v += __shfl_xor(v, 2, 32);
  v += __shfl_xor(v, 4, 32);
  v += __shfl_xor(v, 8, 32);
  return v;
}

// ---------------------------------------------------------------------------
// Kernel 1: q/k/v projections.  q,k stored bf16 [b*S + s][64]; v stored
// transposed bf16 [b][64][S] so attention B-fragments are contiguous.
// One wave = one 16-row M tile; 256 threads/block = 8 M tiles; grid = 64.
// ---------------------------------------------------------------------------
__global__ __launch_bounds__(256) void proj_kernel(
    const float* __restrict__ x,
    const float* __restrict__ Wq, const float* __restrict__ bq,
    const float* __restrict__ Wk, const float* __restrict__ bk,
    const float* __restrict__ Wv, const float* __restrict__ bv,
    unsigned short* __restrict__ qg_, unsigned short* __restrict__ kg_,
    unsigned short* __restrict__ vtg_)
{
  __bf16* qg  = (__bf16*)qg_;
  __bf16* kg  = (__bf16*)kg_;
  __bf16* vtg = (__bf16*)vtg_;

  // W tiles, bf16, transposed: sWt[m][n*32 + kk], 3 x 4KB
  __shared__ alignas(32) __bf16 sWt[3][DH * 32];

  const int tid  = threadIdx.x;
  const int wave = tid >> 5;
  const int lane = tid & 31;
  const int hf   = lane >> 4;      // half-wave: 0 or 1
  const int ln   = lane & 15;
  const int mbase = blockIdx.x * 128;        // flattened (b*S + s) row base
  const int arow  = mbase + wave * 16 + ln;  // A-fragment row for this lane

  const float* Ws[3] = {Wq, Wk, Wv};

  v8f acc[3][4];
#pragma unroll
  for (int m = 0; m < 3; ++m)
#pragma unroll
    for (int t = 0; t < 4; ++t)
#pragma unroll
      for (int r = 0; r < 8; ++r) acc[m][t][r] = 0.0f;

  // Staging decomposition: thread owns column n = tid&63, kk chunk of 8.
  const int st_n   = tid & 63;
  const int st_kk0 = (tid >> 6) * 8;

  for (int kb = 0; kb < DM / 32; ++kb) {
    // Cooperative stage: 32(kk) x 64(n) f32 -> bf16 transposed in LDS.
    // kk-strided dword loads (coalesced across the 64 consecutive-n lanes),
    // one packed ds_store_b128 per matrix.
#pragma unroll
    for (int m = 0; m < 3; ++m) {
      const float* wp = Ws[m] + (size_t)(kb * 32 + st_kk0) * DH + st_n;
      v8bf pk;
#pragma unroll
      for (int j = 0; j < 8; ++j) pk[j] = (__bf16)wp[(size_t)j * DH];
      *(v8bf*)&sWt[m][st_n * 32 + st_kk0] = pk;
    }
    __syncthreads();

    // A fragment: x row (f32 -> bf16). Lane layout (matches B fragment):
    // lanes 0-15: K in {0..15}; lanes 16-31: K in {16..31}.
    v16bf afrag;
    {
      const float* xr = x + (size_t)arow * DM + kb * 32 + hf * 16;
#pragma unroll
      for (int j = 0; j < 8; ++j) {
        afrag[j]     = (__bf16)xr[j];
        afrag[8 + j] = (__bf16)xr[8 + j];
      }
    }

#pragma unroll
    for (int m = 0; m < 3; ++m) {
#pragma unroll
      for (int t = 0; t < 4; ++t) {
        const int n = t * 16 + ln;
        v16bf bfrag = *(const v16bf*)&sWt[m][n * 32 + hf * 16];
        acc[m][t] = wmma_bf16_f32(afrag, bfrag, acc[m][t]);
      }
    }
    __syncthreads();
  }

  // Epilogue: bias add + store. C layout: row = r + 8*hf, col = 16t + ln.
#pragma unroll
  for (int t = 0; t < 4; ++t) {
    const int h = t * 16 + ln;
    const float biq = bq[h], bik = bk[h], biv = bv[h];
#pragma unroll
    for (int r = 0; r < 8; ++r) {
      const int row = mbase + wave * 16 + r + 8 * hf;
      const int b   = row >> 11;          // / S_
      const int s   = row & (S_ - 1);
      qg[(size_t)row * DH + h]           = (__bf16)(acc[0][t][r] + biq);
      kg[(size_t)row * DH + h]           = (__bf16)(acc[1][t][r] + bik);
      vtg[((size_t)b * DH + h) * S_ + s] = (__bf16)(acc[2][t][r] + biv);
    }
  }
}

// ---------------------------------------------------------------------------
// Kernel 2: flash attention with online softmax + online entropy.
// Wave = one 16-row Q tile; block = 8 waves = 128 Q rows; grid = 64.
// K-loop steps 32 keys: 4 score WMMAs + 4 PV WMMAs per wave per step.
// K/V tiles staged with gfx1250 async global->LDS copies when available.
// ---------------------------------------------------------------------------
__global__ __launch_bounds__(256) void attn_kernel(
    const unsigned short* __restrict__ qg_, const unsigned short* __restrict__ kg_,
    const unsigned short* __restrict__ vtg_, const int* __restrict__ mask,
    float* __restrict__ out, float* __restrict__ ent)
{
  const __bf16* qg  = (const __bf16*)qg_;
  const __bf16* kg  = (const __bf16*)kg_;
  const __bf16* vtg = (const __bf16*)vtg_;

  __shared__ alignas(32) __bf16 sK[32 * DH];    // [key][d]   4KB
  __shared__ alignas(32) __bf16 sVt[DH * 32];   // [d][key]   4KB
  __shared__ alignas(32) __bf16 sP[8][16 * 32]; // per-wave P 8KB

  const int tid  = threadIdx.x;
  const int wave = tid >> 5;
  const int lane = tid & 31;
  const int hf   = lane >> 4;
  const int ln   = lane & 15;

  const int qtiles_per_b = S_ / 128;                 // 16
  const int b      = blockIdx.x / qtiles_per_b;
  const int qblock = (blockIdx.x % qtiles_per_b) * 128;
  const int qbase  = qblock + wave * 16;

  // Q fragments (fixed for whole K loop): f=0 -> d 0..31, f=1 -> d 32..63
  v16bf qf[2];
  {
    const __bf16* qr = qg + ((size_t)b * S_ + qbase + ln) * DH;
#pragma unroll
    for (int f = 0; f < 2; ++f) {
#pragma unroll
      for (int j = 0; j < 8; ++j) {
        qf[f][j]     = qr[f * 32 + hf * 16 + j];
        qf[f][8 + j] = qr[f * 32 + hf * 16 + 8 + j];
      }
    }
  }

  float mrow[8], lrow[8], swrow[8];
  v8f o[4];
#pragma unroll
  for (int r = 0; r < 8; ++r) { mrow[r] = -3.0e38f; lrow[r] = 0.f; swrow[r] = 0.f; }
#pragma unroll
  for (int t = 0; t < 4; ++t)
#pragma unroll
    for (int r = 0; r < 8; ++r) o[t][r] = 0.f;

  const size_t mask_row0 = ((size_t)b * S_ + qbase) * S_;

  // Staging decomposition (16B per thread per tile).
  const int st_key = tid >> 3, st_d0 = (tid & 7) * 8;   // K tile
  const int st_d   = tid >> 2, st_c0 = (tid & 3) * 8;   // Vt tile

  for (int kb = 0; kb < S_; kb += 32) {
    // Cooperative stage of K (row-major) and Vt (d-major): async -> LDS.
    copy16_g2lds(kg + ((size_t)b * S_ + kb + st_key) * DH + st_d0,
                 &sK[st_key * DH + st_d0]);
    copy16_g2lds(vtg + ((size_t)b * DH + st_d) * S_ + kb + st_c0,
                 &sVt[st_d * 32 + st_c0]);
    if (kb + 32 < S_)
      __builtin_prefetch((const void*)&mask[mask_row0 + (size_t)(hf * 8) * S_ + kb + 32 + ln], 0, 1);
    async_copy_fence();
    __syncthreads();

    // Scores for two 16-key tiles; K-dim (head dim 64) = 2 WMMA steps each.
    v8f c0v, c1v;
#pragma unroll
    for (int r = 0; r < 8; ++r) { c0v[r] = 0.f; c1v[r] = 0.f; }
#pragma unroll
    for (int f = 0; f < 2; ++f) {
      v16bf kb0 = *(const v16bf*)&sK[(ln) * DH + f * 32 + hf * 16];
      v16bf kb1 = *(const v16bf*)&sK[(16 + ln) * DH + f * 32 + hf * 16];
      c0v = wmma_bf16_f32(qf[f], kb0, c0v);
      c1v = wmma_bf16_f32(qf[f], kb1, c1v);
    }

    // Mask + scale + online softmax/entropy.  Row = r + 8*hf, key col = ln.
    // Mask is 67MB streamed once: non-temporal so it doesn't rinse L2.
#pragma unroll
    for (int r = 0; r < 8; ++r) {
      const int row = r + 8 * hf;
      const size_t mi = mask_row0 + (size_t)row * S_ + kb + ln;
      const int m0 = __builtin_nontemporal_load(mask + mi);
      const int m1 = __builtin_nontemporal_load(mask + mi + 16);
      float s0 = m0 ? c0v[r] * 0.125f : -1.0e9f;
      float s1 = m1 ? c1v[r] * 0.125f : -1.0e9f;
      float tm = red_max16(fmaxf(s0, s1));
      float mn = fmaxf(mrow[r], tm);
      float corr = __expf(mrow[r] - mn);
      mrow[r] = mn;
      float p0 = __expf(s0 - mn), p1 = __expf(s1 - mn);
      float ps = red_sum16(p0 + p1);
      float ss = red_sum16(p0 * s0 + p1 * s1);
      lrow[r]  = lrow[r]  * corr + ps;
      swrow[r] = swrow[r] * corr + ss;
#pragma unroll
      for (int t = 0; t < 4; ++t) o[t][r] *= corr;
      sP[wave][row * 32 + ln]      = (__bf16)p0;
      sP[wave][row * 32 + 16 + ln] = (__bf16)p1;
    }

    // P: C-layout -> A-layout through per-wave LDS (same-wave DS ordering).
    v16bf pf;
#pragma unroll
    for (int j = 0; j < 8; ++j) {
      pf[j]     = sP[wave][ln * 32 + hf * 16 + j];
      pf[8 + j] = sP[wave][ln * 32 + hf * 16 + 8 + j];
    }
#pragma unroll
    for (int t = 0; t < 4; ++t) {
      v16bf vb = *(const v16bf*)&sVt[(t * 16 + ln) * 32 + hf * 16];
      o[t] = wmma_bf16_f32(pf, vb, o[t]);
    }
    __syncthreads();
  }

  // Epilogue: normalize O, emit entropy H = m + log(Z) - sum(p*s)/Z.
#pragma unroll
  for (int t = 0; t < 4; ++t) {
    const int h = t * 16 + ln;
#pragma unroll
    for (int r = 0; r < 8; ++r) {
      const int row = qbase + r + 8 * hf;
      out[((size_t)b * S_ + row) * DH + h] = o[t][r] / lrow[r];
    }
  }
  if (ln == 0) {
#pragma unroll
    for (int r = 0; r < 8; ++r) {
      const int row = qbase + r + 8 * hf;
      ent[(size_t)b * S_ + row] = mrow[r] + __logf(lrow[r]) - swrow[r] / lrow[r];
    }
  }
}

extern "C" void kernel_launch(void* const* d_in, const int* in_sizes, int n_in,
                              void* d_out, int out_size, void* d_ws, size_t ws_size,
                              hipStream_t stream) {
  const float* x  = (const float*)d_in[0];
  const int* mask = (const int*)d_in[1];
  const float* Wq = (const float*)d_in[2];
  const float* bq = (const float*)d_in[3];
  const float* Wk = (const float*)d_in[4];
  const float* bk = (const float*)d_in[5];
  const float* Wv = (const float*)d_in[6];
  const float* bv = (const float*)d_in[7];

  unsigned short* qg  = (unsigned short*)d_ws;                 // bf16 q  [B*S][64]
  unsigned short* kg  = qg + (size_t)B_ * S_ * DH;             // bf16 k  [B*S][64]
  unsigned short* vtg = kg + (size_t)B_ * S_ * DH;             // bf16 vT [B][64][S]

  float* out = (float*)d_out;                  // [B,S,64]
  float* ent = out + (size_t)B_ * S_ * DH;     // [B,S]

  proj_kernel<<<(B_ * S_) / 128, 256, 0, stream>>>(x, Wq, bq, Wk, bk, Wv, bv,
                                                   qg, kg, vtg);
  attn_kernel<<<B_ * (S_ / 128), 256, 0, stream>>>(qg, kg, vtg, mask, out, ent);
}
